// GMFlowMatching_70411693851138
// MI455X (gfx1250) — matrix operand order repaired and checked by reading, not compiled, not run on hardware
//
#include <hip/hip_runtime.h>
#include <hip/hip_bf16.h>
#include <math.h>

typedef __attribute__((ext_vector_type(16))) _Float16 v16h;
typedef __attribute__((ext_vector_type(8)))  float    v8f;

#define BATCH 2
#define CCH   128
#define HH    96
#define WW    96
#define HWSZ  (HH*WW)        // 9216
#define QT    (HWSZ/16)      // 576 tiles of 16 queries/keys
#define KC    4              // 128 / 32 K-chunks per WMMA pass
#define FRAG  512            // halfs per (tile,kc) fragment block: 32 lanes * 16
#define TILE_HALFS (KC*FRAG*4)   // 8192 halfs per 64-key LDS tile (16 KB)
#define TILE_BYTES (TILE_HALFS*2)
#define NT    (HWSZ/64)      // 144 key tiles of 64
// exp(score/ (sqrt(128)*0.1)) == exp2(dot * EXP2C); clamp +-50 is provably dead
// since |dot| <= 1  =>  |exp arg| <= 0.884.
#define EXP2C 1.2751743104f  // (1/(sqrt(128)*0.1)) * log2(e)

// ---------------------------------------------------------------------------
// 1) per-pixel reciprocal L2 norm over channels (eps-clamped like reference)
// ---------------------------------------------------------------------------
__global__ void norm_kernel(const float* __restrict__ fL, const float* __restrict__ fR,
                            float* __restrict__ rnL, float* __restrict__ rnR) {
  int tid = blockIdx.x * blockDim.x + threadIdx.x;
  const int N = BATCH * HWSZ;
  if (tid >= 2 * N) return;
  int side = tid / N;
  int r    = tid - side * N;
  int b    = r / HWSZ;
  int q    = r - b * HWSZ;
  const float* src = side ? fR : fL;
  const float* p = src + (size_t)b * CCH * HWSZ + q;
  float sum = 0.f;
  #pragma unroll 4
  for (int c = 0; c < CCH; ++c) {
    float v = p[(size_t)c * HWSZ];
    sum += v * v;
  }
  float rn = 1.f / fmaxf(sqrtf(sum), 1e-6f);
  (side ? rnR : rnL)[r] = rn;
}

// ---------------------------------------------------------------------------
// 2) normalize + convert to f16, packed into WMMA fragment layouts.
//    A (16x32 f16, ISA 7.12.2): lane m=L&15, half=L>>4;
//      vgpr v, elem p: K = 2*(v&3) + 16*(v>>2) + 8*half + p
//    B (32x16 f16): lane n=L&15, half=L>>4; elem i: K = (i&7) + 8*half + 16*(i>>3)
// ---------------------------------------------------------------------------
__global__ void pack_kernel(const float* __restrict__ fL, const float* __restrict__ fR,
                            const float* __restrict__ rnL, const float* __restrict__ rnR,
                            _Float16* __restrict__ Apack, _Float16* __restrict__ Bpack) {
  int tid = blockIdx.x * blockDim.x + threadIdx.x;
  const int per_side = BATCH * QT * KC * 32;
  if (tid >= 2 * per_side) return;
  int side = tid / per_side;
  int r    = tid - side * per_side;
  int lane = r & 31;
  int kc   = (r >> 5) & 3;
  int tile = (r >> 7) % QT;
  int b    = (r >> 7) / QT;
  const float* src = side ? fR : fL;
  const float* rn  = side ? rnR : rnL;
  _Float16* dst    = side ? Bpack : Apack;
  int half   = lane >> 4;
  int rowcol = lane & 15;
  int pix    = tile * 16 + rowcol;          // query row (A) or key column (B)
  float scale = rn[b * HWSZ + pix];
  const float* sp = src + (size_t)b * CCH * HWSZ + pix;

  _Float16 vals[16];
  if (side == 0) {
    #pragma unroll
    for (int i = 0; i < 16; ++i) {
      int v = i >> 1, p = i & 1;
      int K = 2 * (v & 3) + 16 * (v >> 2) + 8 * half + p;
      int c = kc * 32 + K;
      vals[i] = (_Float16)(sp[(size_t)c * HWSZ] * scale);
    }
  } else {
    #pragma unroll
    for (int i = 0; i < 16; ++i) {
      int K = (i & 7) + 8 * half + 16 * (i >> 3);
      int c = kc * 32 + K;
      vals[i] = (_Float16)(sp[(size_t)c * HWSZ] * scale);
    }
  }
  size_t off = ((size_t)(b * QT + tile) * KC + kc) * FRAG + (size_t)lane * 16;
  #pragma unroll
  for (int i = 0; i < 16; ++i) dst[off + i] = vals[i];
}

// ---------------------------------------------------------------------------
// async global -> LDS staging (gfx1250 GLOBAL_LOAD_ASYNC_TO_LDS_B128).
// GVS mode: LDS address from vdst, global address from saddr + vaddr, both plus offs.
// Each thread moves 128 B (8 x b128). Tracked by ASYNCcnt.
// ---------------------------------------------------------------------------
__device__ __forceinline__ void async_copy_tile(unsigned lds_base_b, unsigned goff_b,
                                                unsigned long long gbase) {
#define ASYNC_B128(OFF)                                                        \
  asm volatile("global_load_async_to_lds_b128 %0, %1, %2 offset:" #OFF         \
               :: "v"(lds_base_b), "v"(goff_b), "s"(gbase) : "memory")
  ASYNC_B128(0);  ASYNC_B128(16); ASYNC_B128(32); ASYNC_B128(48);
  ASYNC_B128(64); ASYNC_B128(80); ASYNC_B128(96); ASYNC_B128(112);
#undef ASYNC_B128
}
__device__ __forceinline__ void wait_async_all() {
  asm volatile("s_wait_asynccnt 0x0" ::: "memory");
}

// ---------------------------------------------------------------------------
// 3) fused correlation + softmax-expectation (flash-style, fixed exp base).
// ---------------------------------------------------------------------------
__device__ __forceinline__ v8f score_tile16(v16h a0, v16h a1, v16h a2, v16h a3,
                                            const v16h* __restrict__ Bp) {
  v8f acc = {};
  acc = __builtin_amdgcn_wmma_f32_16x16x32_f16(false, a0, false, Bp[0],  (short)0, acc, false, false);
  acc = __builtin_amdgcn_wmma_f32_16x16x32_f16(false, a1, false, Bp[32], (short)0, acc, false, false);
  acc = __builtin_amdgcn_wmma_f32_16x16x32_f16(false, a2, false, Bp[64], (short)0, acc, false, false);
  acc = __builtin_amdgcn_wmma_f32_16x16x32_f16(false, a3, false, Bp[96], (short)0, acc, false, false);
  return acc;
}

__global__ __launch_bounds__(128)
void corr_softmax_kernel(const _Float16* __restrict__ Apack,
                         const _Float16* __restrict__ Bpack,
                         float* __restrict__ out) {
  __shared__ _Float16 sB[2][TILE_HALFS];          // 2 x 16 KB double buffer

  const int lane = threadIdx.x & 31;
  const int wv   = threadIdx.x >> 5;
  const int t    = blockIdx.x * 4 + wv;           // global q-tile id
  const int b    = t / QT;                        // same for all 4 waves (QT%4==0)
  const int qt   = t - b * QT;

  // A fragments for this q-tile (K = 0..127), resident for whole kernel
  const v16h* Ap = (const v16h*)Apack + (size_t)(b * QT + qt) * (KC * 32) + lane;
  v16h a0 = Ap[0], a1 = Ap[32], a2 = Ap[64], a3 = Ap[96];

  const _Float16* Bbat = Bpack + (size_t)b * QT * (KC * FRAG);

  // per-thread staging addresses (flat LDS aperture: low 32 bits = LDS address)
  const unsigned goff   = (unsigned)threadIdx.x * 128u;
  const unsigned lds0   = (unsigned)(unsigned long long)(uintptr_t)&sB[0][0] + goff;
  const unsigned lds1   = (unsigned)(unsigned long long)(uintptr_t)&sB[1][0] + goff;

  // per-lane private accumulators (no cross-lane work in the hot loop)
  float l[8], sx[8], sy[8], Mx[8];
  #pragma unroll
  for (int j = 0; j < 8; ++j) { l[j] = 0.f; sx[j] = 0.f; sy[j] = 0.f; Mx[j] = -1e30f; }

  // incremental key coordinates for this lane's 4 columns (k = kt*64 + nsub*16 + n)
  const int n = lane & 15;
  float x0 = (float)n,        y0 = 0.f;
  float x1 = (float)(n + 16), y1 = 0.f;
  float x2 = (float)(n + 32), y2 = 0.f;
  float x3 = (float)(n + 48), y3 = 0.f;

  // preload tile 0 into buffer 0
  async_copy_tile(lds0, goff, (unsigned long long)(uintptr_t)Bbat);
  wait_async_all();
  __syncthreads();

  for (int kt = 0; kt < NT; ++kt) {
    const int cur = kt & 1;

    // kick off next tile's async copy into the other buffer; safe, because the
    // barrier ending iteration kt-1 retired all readers of that buffer.
    if (kt + 1 < NT) {
      async_copy_tile(cur ? lds0 : lds1, goff,
                      (unsigned long long)(uintptr_t)(Bbat + (size_t)(kt + 1) * TILE_HALFS));
    }

    // 16 WMMAs: 16 queries x 64 keys, K = 128
    const v16h* Bt = (const v16h*)&sB[cur][0] + lane;
    v8f acc0 = score_tile16(a0, a1, a2, a3, Bt);
    v8f acc1 = score_tile16(a0, a1, a2, a3, Bt + 128);
    v8f acc2 = score_tile16(a0, a1, a2, a3, Bt + 256);
    v8f acc3 = score_tile16(a0, a1, a2, a3, Bt + 384);

    #pragma unroll
    for (int j = 0; j < 8; ++j) {
      float t0 = acc0[j], t1 = acc1[j], t2 = acc2[j], t3 = acc3[j];
      Mx[j] = fmaxf(Mx[j], fmaxf(fmaxf(t0, t1), fmaxf(t2, t3)));
      float p0 = __builtin_amdgcn_exp2f(t0 * EXP2C);
      float p1 = __builtin_amdgcn_exp2f(t1 * EXP2C);
      float p2 = __builtin_amdgcn_exp2f(t2 * EXP2C);
      float p3 = __builtin_amdgcn_exp2f(t3 * EXP2C);
      l[j]  += (p0 + p1) + (p2 + p3);
      sx[j] += p0 * x0 + p1 * x1 + p2 * x2 + p3 * x3;
      sy[j] += p0 * y0 + p1 * y1 + p2 * y2 + p3 * y3;
    }

    // advance coordinates by 64 keys (x < 96 always, one conditional wrap)
    x0 += 64.f; if (x0 >= 96.f) { x0 -= 96.f; y0 += 1.f; }
    x1 += 64.f; if (x1 >= 96.f) { x1 -= 96.f; y1 += 1.f; }
    x2 += 64.f; if (x2 >= 96.f) { x2 -= 96.f; y2 += 1.f; }
    x3 += 64.f; if (x3 >= 96.f) { x3 -= 96.f; y3 += 1.f; }

    wait_async_all();                // own wave's async writes landed in LDS
    __syncthreads();                 // all waves: reads of cur done, next tile visible
  }

  // one-time cross-lane reduction within each 16-lane half-group
  const int half = lane >> 4;
  #pragma unroll
  for (int j = 0; j < 8; ++j) {
    float lv = l[j], sxv = sx[j], syv = sy[j], mv = Mx[j];
    #pragma unroll
    for (int mask = 1; mask <= 8; mask <<= 1) {
      lv  += __shfl_xor(lv,  mask);
      sxv += __shfl_xor(sxv, mask);
      syv += __shfl_xor(syv, mask);
      mv   = fmaxf(mv, __shfl_xor(mv, mask));
    }
    if ((lane & 15) == 0) {
      int q = qt * 16 + j + 8 * half;          // C/D layout: rows j (lanes 0-15), j+8 (16-31)
      float inv = 1.f / lv;
      float ex = sxv * inv, ey = syv * inv;
      float qx = (float)(q % WW), qy = (float)(q / WW);
      out[(size_t)b * 2 * HWSZ + q]            = ex - qx;           // flow x
      out[(size_t)b * 2 * HWSZ + HWSZ + q]     = ey - qy;           // flow y
      out[(size_t)BATCH * 2 * HWSZ + (size_t)b * HWSZ + q] =
          __builtin_amdgcn_exp2f(mv * EXP2C) * inv;                 // conf = max prob
    }
  }
}

// ---------------------------------------------------------------------------
extern "C" void kernel_launch(void* const* d_in, const int* in_sizes, int n_in,
                              void* d_out, int out_size, void* d_ws, size_t ws_size,
                              hipStream_t stream) {
  const float* fL = (const float*)d_in[0];
  const float* fR = (const float*)d_in[1];
  float* out = (float*)d_out;

  char* ws = (char*)d_ws;
  float* rnL = (float*)ws;                             // BATCH*HWSZ floats
  float* rnR = rnL + BATCH * HWSZ;                     // BATCH*HWSZ floats
  _Float16* Apack = (_Float16*)(rnR + BATCH * HWSZ);   // ~4.7 MB
  _Float16* Bpack = Apack + (size_t)BATCH * QT * KC * FRAG;

  {
    int N = 2 * BATCH * HWSZ;
    norm_kernel<<<(N + 255) / 256, 256, 0, stream>>>(fL, fR, rnL, rnR);
  }
  {
    int N = 2 * BATCH * QT * KC * 32;
    pack_kernel<<<(N + 255) / 256, 256, 0, stream>>>(fL, fR, rnL, rnR, Apack, Bpack);
  }
  {
    int ntiles = BATCH * QT;                           // 1152 q-tiles, 4 waves/block
    corr_softmax_kernel<<<ntiles / 4, 128, 0, stream>>>(Apack, Bpack, out);
  }
}
